// NeuralMemory_7473243095581
// MI455X (gfx1250) — compile-verified, matrix operand order checked by reading, not run on hardware
//
#include <hip/hip_runtime.h>
#include <hip/hip_bf16.h>
#include <cstdint>
#include <cstddef>

typedef __attribute__((ext_vector_type(16))) _Float16 v16h;
typedef __attribute__((ext_vector_type(8)))  _Float16 v8h;
typedef __attribute__((ext_vector_type(8)))  float    v8f;
typedef __attribute__((ext_vector_type(4)))  int      v4i;

#define B_  2
#define T_  2048
#define D_  256
#define DH_ 1024
#define CC_ 64
#define NC_ (T_/CC_)
#define BT_ (B_*T_)

#if __has_builtin(__builtin_amdgcn_global_load_async_to_lds_b128)
#define HAVE_ASYNC_LDS 1
#else
#define HAVE_ASYNC_LDS 0
#endif

__device__ __forceinline__ float sigf(float x) { return 1.0f / (1.0f + __expf(-x)); }

#if HAVE_ASYNC_LDS
typedef __attribute__((address_space(1))) v4i* gv4i_p;
typedef __attribute__((address_space(3))) v4i* lv4i_p;
__device__ __forceinline__ void cp16a(const _Float16* g, _Float16* l) {
  __builtin_amdgcn_global_load_async_to_lds_b128((gv4i_p)g, (lv4i_p)l, 0, 0);
}
__device__ __forceinline__ void wait_async0() {
#if __has_builtin(__builtin_amdgcn_s_wait_asynccnt)
  __builtin_amdgcn_s_wait_asynccnt(0);
#else
  asm volatile("s_wait_asynccnt 0" ::: "memory");
#endif
}
#endif

// Load one 16x32 (or 32x16) f16 WMMA fragment from a row-major LDS tile
// (row stride 32 f16). Per ISA 7.12.2: elements 0..7 hold K = lh+0..7,
// elements 8..15 hold K = 16+lh+0..7, lh = (lane>=16) ? 8 : 0.
__device__ __forceinline__ v16h ldfrag(const _Float16* p, int lh) {
  v8h lo = *(const v8h*)(p + lh);
  v8h hi = *(const v8h*)(p + lh + 16);
  v16h r;
#pragma unroll
  for (int i = 0; i < 8; ++i) { r[i] = lo[i]; r[i + 8] = hi[i]; }
  return r;
}

// One wave k-step: 32(M)x64(N) worth of WMMAs from LDS tiles (row stride 32).
__device__ __forceinline__ void wmma_block(const _Float16* Asb, const _Float16* Bsb,
                                           int wm, int wn, int l15, int lh,
                                           v8f (&acc)[2][4]) {
  v16h a0 = ldfrag(&Asb[(wm + l15) * 32], lh);
  v16h a1 = ldfrag(&Asb[(wm + 16 + l15) * 32], lh);
#pragma unroll
  for (int nt = 0; nt < 4; ++nt) {
    v16h b = ldfrag(&Bsb[(wn + nt * 16 + l15) * 32], lh);
    acc[0][nt] = __builtin_amdgcn_wmma_f32_16x16x32_f16(
        false, a0, false, b, (short)0, acc[0][nt], false, false);
    acc[1][nt] = __builtin_amdgcn_wmma_f32_16x16x32_f16(
        false, a1, false, b, (short)0, acc[1][nt], false, false);
  }
}

struct GemmP {
  const _Float16* A;  long sA; int lda;     // A: M x K, K contiguous
  const _Float16* Bt; long sB; int ldb;     // Bt: N x K, K contiguous (B transposed)
  int M, N, K;
  float* out; long sO; int ldo;             // EPI 0/1
  const float* bias;                        // EPI 2
  float* rowacc; long sRow; float invN;     // EPI 2 (atomic row mean of sigmoid)
  _Float16* out16; int ldo16;               // EPI 3 (silu -> f16)
  _Float16* outT;                           // EPI 3 optional transposed [b, n, t]
  const float* v; long sV; int ldv;         // EPI 4 (res = pred - v)
  const float* wS; const float* wF; long sW;// EPI 4 per-row (time) coefficients
  _Float16* r1T; _Float16* r2T; long sR;    // EPI 4 outputs [n*CC_ + m], f16
};

// Generic WMMA GEMM: block tile 64(M) x 128(N), 4 waves as 2(M) x 2(N);
// wave computes 32x64; K step 32; double-buffered LDS with async
// global->LDS prefetch (or register-staged prefetch fallback).
template <int EPI>
__launch_bounds__(128)
__global__ void gemm_epi(GemmP p) {
  __shared__ __align__(16) _Float16 As[2][64 * 32];
  __shared__ __align__(16) _Float16 Bs[2][128 * 32];
  const int z  = blockIdx.z;
  const int n0 = blockIdx.x * 128, m0 = blockIdx.y * 64;
  const int tid = threadIdx.x, lane = tid & 31, w = tid >> 5;
  const int wm = (w >> 1) * 32;   // wave row offset within tile
  const int wn = (w & 1) * 64;    // wave col offset within tile
  const int lh = (lane & 16) ? 8 : 0;
  const int l15 = lane & 15;
  const _Float16* Ab = p.A  + (long)z * p.sA;
  const _Float16* Bb = p.Bt + (long)z * p.sB;
  v8f acc[2][4] = {};

  // staging assignment: thread covers A row r (segment h) and B rows r, r+64
  const int r = tid >> 1, h = tid & 1;
  const long aoff  = (long)(m0 + r) * p.lda + h * 16;
  const long boff0 = (long)(n0 + r) * p.ldb + h * 16;
  const long boff1 = (long)(n0 + 64 + r) * p.ldb + h * 16;
  const int loA  = r * 32 + h * 16;
  const int loB0 = r * 32 + h * 16;
  const int loB1 = (64 + r) * 32 + h * 16;

#if HAVE_ASYNC_LDS
#define GSTAGE(k0, bi) do { \
    cp16a(Ab + aoff  + (k0),     &As[bi][loA]);      \
    cp16a(Ab + aoff  + (k0) + 8, &As[bi][loA + 8]);  \
    cp16a(Bb + boff0 + (k0),     &Bs[bi][loB0]);     \
    cp16a(Bb + boff0 + (k0) + 8, &Bs[bi][loB0 + 8]); \
    cp16a(Bb + boff1 + (k0),     &Bs[bi][loB1]);     \
    cp16a(Bb + boff1 + (k0) + 8, &Bs[bi][loB1 + 8]); \
  } while (0)
  GSTAGE(0, 0);
  wait_async0();
  __syncthreads();
  int bi = 0;
  for (int k0 = 0; k0 < p.K; k0 += 32, bi ^= 1) {
    if (k0 + 32 < p.K) GSTAGE(k0 + 32, bi ^ 1);
    wmma_block(As[bi], Bs[bi], wm, wn, l15, lh, acc);
    wait_async0();
    __syncthreads();
  }
#undef GSTAGE
#else
  {
    v8h a0 = *(const v8h*)(Ab + aoff),      a1 = *(const v8h*)(Ab + aoff + 8);
    v8h b0 = *(const v8h*)(Bb + boff0),     b1 = *(const v8h*)(Bb + boff0 + 8);
    v8h b2 = *(const v8h*)(Bb + boff1),     b3 = *(const v8h*)(Bb + boff1 + 8);
    *(v8h*)&As[0][loA] = a0;      *(v8h*)&As[0][loA + 8] = a1;
    *(v8h*)&Bs[0][loB0] = b0;     *(v8h*)&Bs[0][loB0 + 8] = b1;
    *(v8h*)&Bs[0][loB1] = b2;     *(v8h*)&Bs[0][loB1 + 8] = b3;
  }
  __syncthreads();
  int bi = 0;
  for (int k0 = 0; k0 < p.K; k0 += 32, bi ^= 1) {
    const bool more = (k0 + 32) < p.K;
    v8h a0, a1, b0, b1, b2, b3;
    if (more) {  // issue next-tile loads before compute (overlap latency)
      a0 = *(const v8h*)(Ab + aoff  + k0 + 32);
      a1 = *(const v8h*)(Ab + aoff  + k0 + 40);
      b0 = *(const v8h*)(Bb + boff0 + k0 + 32);
      b1 = *(const v8h*)(Bb + boff0 + k0 + 40);
      b2 = *(const v8h*)(Bb + boff1 + k0 + 32);
      b3 = *(const v8h*)(Bb + boff1 + k0 + 40);
    }
    wmma_block(As[bi], Bs[bi], wm, wn, l15, lh, acc);
    if (more) {
      *(v8h*)&As[bi ^ 1][loA] = a0;      *(v8h*)&As[bi ^ 1][loA + 8] = a1;
      *(v8h*)&Bs[bi ^ 1][loB0] = b0;     *(v8h*)&Bs[bi ^ 1][loB0 + 8] = b1;
      *(v8h*)&Bs[bi ^ 1][loB1] = b2;     *(v8h*)&Bs[bi ^ 1][loB1 + 8] = b3;
    }
    __syncthreads();
  }
#endif

  // ---- epilogue ----
#pragma unroll
  for (int ri = 0; ri < 2; ++ri) {
    const int mrBase = m0 + wm + ri * 16 + ((lane & 16) ? 8 : 0);
#pragma unroll
    for (int nt = 0; nt < 4; ++nt) {
      const int n = n0 + wn + nt * 16 + l15;
#pragma unroll
      for (int rr = 0; rr < 8; ++rr) {
        const int m = mrBase + rr;
        float val = acc[ri][nt][rr];
        if constexpr (EPI == 0) {
          p.out[(long)z * p.sO + (long)m * p.ldo + n] = val;
        } else if constexpr (EPI == 1) {
          p.out[(long)z * p.sO + (long)m * p.ldo + n] = sigf(val);
        } else if constexpr (EPI == 2) {
          float s = sigf(val + p.bias[n]);
          atomicAdd(p.rowacc + (long)z * p.sRow + m, s * p.invN);
        } else if constexpr (EPI == 3) {
          float s = val * sigf(val);
          p.out16[(long)m * p.ldo16 + n] = (_Float16)s;
          if (p.outT) {
            int b = m / T_, t = m - b * T_;
            p.outT[((long)b * DH_ + n) * T_ + t] = (_Float16)s;
          }
        } else if constexpr (EPI == 4) {
          float res = val - p.v[(long)z * p.sV + (long)m * p.ldv + n];
          float cs = p.wS[(long)z * p.sW + m];
          float cf = p.wF[(long)z * p.sW + m];
          p.r1T[(long)z * p.sR + (long)n * CC_ + m] = (_Float16)(cs * res);
          p.r2T[(long)z * p.sR + (long)n * CC_ + m] = (_Float16)(cf * res);
        }
      }
    }
  }
}

// Fused dual-A state update: S_new = e*S + r1T@hk ; fW_new = b*fW + s*S + r2T@hk.
// M = D tile 64, N = DH tile 64, K = CC_ (2 k-steps, fully pipelined into the
// two LDS buffers). Wave computes 16(M) x 64(N) for both accumulator sets.
__launch_bounds__(128)
__global__ void state_update(const _Float16* r1T, const _Float16* r2T, long sR,
                             const _Float16* hkT, long sHkT, int tOff,
                             float* S, float* fW, _Float16* fW16,
                             const float* eC, const float* bC, const float* sSc,
                             int scStride) {
  __shared__ __align__(16) _Float16 A1s[2][64 * 32];
  __shared__ __align__(16) _Float16 A2s[2][64 * 32];
  __shared__ __align__(16) _Float16 Bs[2][64 * 32];
  const int z  = blockIdx.z;
  const int n0 = blockIdx.x * 64;   // DH
  const int m0 = blockIdx.y * 64;   // D
  const int tid = threadIdx.x, lane = tid & 31, w = tid >> 5;
  const int lh = (lane & 16) ? 8 : 0, l15 = lane & 15;
  v8f aS[4] = {}, aF[4] = {};
  const int r = tid >> 1, h = tid & 1;
  const long o1 = (long)z * sR   + (long)(m0 + r) * CC_ + h * 16;
  const long o2 = o1;  // same layout for r2T
  const long ob = (long)z * sHkT + (long)(n0 + r) * T_ + tOff + h * 16;
  const int lo = r * 32 + h * 16;

#if HAVE_ASYNC_LDS
#define SSTAGE(k0, bi) do { \
    cp16a(r1T + o1 + (k0),     &A1s[bi][lo]);     \
    cp16a(r1T + o1 + (k0) + 8, &A1s[bi][lo + 8]); \
    cp16a(r2T + o2 + (k0),     &A2s[bi][lo]);     \
    cp16a(r2T + o2 + (k0) + 8, &A2s[bi][lo + 8]); \
    cp16a(hkT + ob + (k0),     &Bs[bi][lo]);      \
    cp16a(hkT + ob + (k0) + 8, &Bs[bi][lo + 8]);  \
  } while (0)
  SSTAGE(0, 0);
  SSTAGE(32, 1);
  wait_async0();
  __syncthreads();
#undef SSTAGE
#else
  {
    v8h x0 = *(const v8h*)(r1T + o1),      x1 = *(const v8h*)(r1T + o1 + 8);
    v8h x2 = *(const v8h*)(r2T + o2),      x3 = *(const v8h*)(r2T + o2 + 8);
    v8h x4 = *(const v8h*)(hkT + ob),      x5 = *(const v8h*)(hkT + ob + 8);
    v8h y0 = *(const v8h*)(r1T + o1 + 32), y1 = *(const v8h*)(r1T + o1 + 40);
    v8h y2 = *(const v8h*)(r2T + o2 + 32), y3 = *(const v8h*)(r2T + o2 + 40);
    v8h y4 = *(const v8h*)(hkT + ob + 32), y5 = *(const v8h*)(hkT + ob + 40);
    *(v8h*)&A1s[0][lo] = x0; *(v8h*)&A1s[0][lo + 8] = x1;
    *(v8h*)&A2s[0][lo] = x2; *(v8h*)&A2s[0][lo + 8] = x3;
    *(v8h*)&Bs[0][lo]  = x4; *(v8h*)&Bs[0][lo + 8]  = x5;
    *(v8h*)&A1s[1][lo] = y0; *(v8h*)&A1s[1][lo + 8] = y1;
    *(v8h*)&A2s[1][lo] = y2; *(v8h*)&A2s[1][lo + 8] = y3;
    *(v8h*)&Bs[1][lo]  = y4; *(v8h*)&Bs[1][lo + 8]  = y5;
  }
  __syncthreads();
#endif

#pragma unroll
  for (int it = 0; it < 2; ++it) {
    v16h a1 = ldfrag(&A1s[it][(w * 16 + l15) * 32], lh);
    v16h a2 = ldfrag(&A2s[it][(w * 16 + l15) * 32], lh);
#pragma unroll
    for (int nt = 0; nt < 4; ++nt) {
      v16h b = ldfrag(&Bs[it][(nt * 16 + l15) * 32], lh);
      aS[nt] = __builtin_amdgcn_wmma_f32_16x16x32_f16(false, a1, false, b, (short)0, aS[nt], false, false);
      aF[nt] = __builtin_amdgcn_wmma_f32_16x16x32_f16(false, a2, false, b, (short)0, aF[nt], false, false);
    }
  }

  const float e  = eC[(long)z * scStride];
  const float bb = bC[(long)z * scStride];
  const float ss = sSc[(long)z * scStride];
  const int mrBase = m0 + w * 16 + ((lane & 16) ? 8 : 0);
#pragma unroll
  for (int nt = 0; nt < 4; ++nt) {
    const int n = n0 + nt * 16 + l15;
#pragma unroll
    for (int rr = 0; rr < 8; ++rr) {
      const int m = mrBase + rr;
      long idx = ((long)z * D_ + m) * DH_ + n;
      float sOld = S[idx], fOld = fW[idx];
      float sNew = e * sOld + aS[nt][rr];
      float fNew = bb * fOld + ss * sOld + aF[nt][rr];
      S[idx] = sNew;
      fW[idx] = fNew;
      fW16[idx] = (_Float16)fNew;
    }
  }
}

// Depthwise conv (k=3, pad 1) + optional l2norm; f16 out if norm, f32 otherwise.
__launch_bounds__(256)
__global__ void conv_kernel(const float* zin, const float* w3,
                            float* outF, _Float16* outH, int doNorm) {
  __shared__ float red[256];
  const int bi = blockIdx.x;            // b*T + t
  const int t  = bi % T_;
  const int d  = threadIdx.x;
  const long base = (long)bi * D_;
  float zm = (t > 0)      ? zin[base - D_ + d] : 0.f;
  float zc = zin[base + d];
  float zp = (t < T_ - 1) ? zin[base + D_ + d] : 0.f;
  float y = w3[d * 3 + 0] * zm + w3[d * 3 + 1] * zc + w3[d * 3 + 2] * zp;
  if (doNorm) {
    red[d] = y * y;
    __syncthreads();
    for (int s = 128; s > 0; s >>= 1) {
      if (d < s) red[d] += red[d + s];
      __syncthreads();
    }
    float nrm = fmaxf(sqrtf(red[0]), 1e-12f);
    outH[base + d] = (_Float16)(y / nrm);
  } else {
    outF[base + d] = y;
  }
}

// Per-(batch, chunk) scalar decay coefficients (collapses both associative scans).
__launch_bounds__(CC_)
__global__ void coef_kernel(const float* theta, const float* eta, const float* alpha,
                            float* wS, float* wF, float* eC, float* bC, float* sS) {
  __shared__ float th[CC_], et[CC_], be[CC_], Pe[CC_], g[CC_];
  __shared__ float scal[3];
  const int bi = blockIdx.x;            // b*NC_ + c
  const int b = bi / NC_, c = bi % NC_;
  const int s = threadIdx.x;
  const long off = (long)b * T_ + (long)c * CC_;
  th[s] = theta[off + s];
  et[s] = eta[off + s];
  be[s] = 1.f - alpha[off + s];
  __syncthreads();
  if (s == 0) {
    float p = 1.f;
    for (int i = CC_ - 1; i >= 0; --i) { Pe[i] = p; p *= et[i]; }
    scal[0] = p;                        // eC = prod eta
    p = 1.f;
    for (int i = CC_ - 1; i >= 0; --i) { g[i] = p; p *= be[i]; }
    scal[1] = p;                        // bC = prod beta
    p = 1.f; float accS = 0.f;
    for (int i = 0; i < CC_; ++i) { p *= et[i]; accS += g[i] * p; }
    scal[2] = accS;                     // sS = sum g_t * A_t
  }
  __syncthreads();
  float ws = -th[s] * Pe[s];
  float pe = 1.f, accF = 0.f;
  for (int t2 = s; t2 < CC_; ++t2) {
    if (t2 > s) pe *= et[t2];
    accF += g[t2] * pe;
  }
  float wf = -th[s] * accF;
  const long wo = ((long)b * NC_ + c) * CC_ + s;
  wS[wo] = ws;
  wF[wo] = wf;
  if (s == 0) { eC[bi] = scal[0]; bC[bi] = scal[1]; sS[bi] = scal[2]; }
}

// LayerNorm over D + sigmoid gate multiply, f16 out for final projection.
__launch_bounds__(256)
__global__ void ln_gate(const float* ret, const float* gate,
                        const float* g, const float* bet, _Float16* y) {
  __shared__ float r1[256], r2[256];
  const int bi = blockIdx.x;
  const int d = threadIdx.x;
  const long base = (long)bi * D_;
  float v = ret[base + d];
  r1[d] = v;
  r2[d] = v * v;
  __syncthreads();
  for (int s = 128; s > 0; s >>= 1) {
    if (d < s) { r1[d] += r1[d + s]; r2[d] += r2[d + s]; }
    __syncthreads();
  }
  float mu = r1[0] * (1.f / D_);
  float var = r2[0] * (1.f / D_) - mu * mu;
  float rstd = rsqrtf(var + 1e-5f);
  float o = (v - mu) * rstd * g[d] + bet[d];
  o *= gate[base + d];
  y[base + d] = (_Float16)o;
}

__global__ void cast_f16(const float* in, _Float16* out, long n) {
  long i = (long)blockIdx.x * blockDim.x + threadIdx.x;
  if (i < n) out[i] = (_Float16)in[i];
}

__global__ void init_state(const float* W2, float* S, float* fW, _Float16* fW16) {
  long i = (long)blockIdx.x * blockDim.x + threadIdx.x;
  const long n = (long)B_ * D_ * DH_;
  if (i < n) {
    float w = W2[i % ((long)D_ * DH_)];
    S[i] = 0.f;
    fW[i] = w;
    fW16[i] = (_Float16)w;
  }
}

__global__ void zero_f32(float* p, long n) {
  long i = (long)blockIdx.x * blockDim.x + threadIdx.x;
  if (i < n) p[i] = 0.f;
}

extern "C" void kernel_launch(void* const* d_in, const int* in_sizes, int n_in,
                              void* d_out, int out_size, void* d_ws, size_t ws_size,
                              hipStream_t stream) {
  (void)in_sizes; (void)n_in; (void)out_size; (void)ws_size;

  const float* x      = (const float*)d_in[0];
  const float* W_K    = (const float*)d_in[1];
  const float* W_V    = (const float*)d_in[2];
  const float* W_Q    = (const float*)d_in[3];
  const float* conv_k = (const float*)d_in[4];
  const float* conv_v = (const float*)d_in[5];
  const float* conv_q = (const float*)d_in[6];
  const float* W_th   = (const float*)d_in[7];
  const float* b_th   = (const float*)d_in[8];
  const float* W_et   = (const float*)d_in[9];
  const float* b_et   = (const float*)d_in[10];
  const float* W_al   = (const float*)d_in[11];
  const float* b_al   = (const float*)d_in[12];
  const float* W1     = (const float*)d_in[13];
  const float* W2     = (const float*)d_in[14];
  const float* ln_g   = (const float*)d_in[15];
  const float* ln_b   = (const float*)d_in[16];
  const float* W_gate = (const float*)d_in[17];
  const float* W_proj = (const float*)d_in[18];

  char* wp = (char*)d_ws;
  auto alloc = [&](size_t bytes) -> char* {
    char* r = wp;
    wp += (bytes + 255) & ~(size_t)255;
    return r;
  };

  const long nBT_D  = (long)BT_ * D_;
  const long nBT_DH = (long)BT_ * DH_;
  const long nDD    = (long)D_ * D_;
  const long nState = (long)B_ * D_ * DH_;

  _Float16* xh   = (_Float16*)alloc(sizeof(_Float16) * nBT_D);
  _Float16* WKh  = (_Float16*)alloc(sizeof(_Float16) * nDD);
  _Float16* WVh  = (_Float16*)alloc(sizeof(_Float16) * nDD);
  _Float16* WQh  = (_Float16*)alloc(sizeof(_Float16) * nDD);
  _Float16* Wthh = (_Float16*)alloc(sizeof(_Float16) * nDD);
  _Float16* Weth = (_Float16*)alloc(sizeof(_Float16) * nDD);
  _Float16* Walh = (_Float16*)alloc(sizeof(_Float16) * nDD);
  _Float16* Wgh  = (_Float16*)alloc(sizeof(_Float16) * nDD);
  _Float16* Wph  = (_Float16*)alloc(sizeof(_Float16) * nDD);
  _Float16* W1h  = (_Float16*)alloc(sizeof(_Float16) * (long)DH_ * D_);
  float* kpre = (float*)alloc(sizeof(float) * nBT_D);
  float* vpre = (float*)alloc(sizeof(float) * nBT_D);
  float* qpre = (float*)alloc(sizeof(float) * nBT_D);
  _Float16* kh = (_Float16*)alloc(sizeof(_Float16) * nBT_D);
  _Float16* qh = (_Float16*)alloc(sizeof(_Float16) * nBT_D);
  float* vF    = (float*)alloc(sizeof(float) * nBT_D);
  float* accs  = (float*)alloc(sizeof(float) * 3L * BT_);
  float* thetaA = accs, *etaA = accs + BT_, *alphaA = accs + 2L * BT_;
  float* gateS = (float*)alloc(sizeof(float) * nBT_D);
  _Float16* hk  = (_Float16*)alloc(sizeof(_Float16) * nBT_DH);
  _Float16* hq  = (_Float16*)alloc(sizeof(_Float16) * nBT_DH);
  _Float16* hkT = (_Float16*)alloc(sizeof(_Float16) * nBT_DH);
  float* wSb = (float*)alloc(sizeof(float) * (long)B_ * NC_ * CC_);
  float* wFb = (float*)alloc(sizeof(float) * (long)B_ * NC_ * CC_);
  float* eCb = (float*)alloc(sizeof(float) * (long)B_ * NC_);
  float* bCb = (float*)alloc(sizeof(float) * (long)B_ * NC_);
  float* sSb = (float*)alloc(sizeof(float) * (long)B_ * NC_);
  float* Sb  = (float*)alloc(sizeof(float) * nState);
  float* fWb = (float*)alloc(sizeof(float) * nState);
  _Float16* fW16b = (_Float16*)alloc(sizeof(_Float16) * nState);
  _Float16* r1Tb  = (_Float16*)alloc(sizeof(_Float16) * (long)B_ * D_ * CC_);
  _Float16* r2Tb  = (_Float16*)alloc(sizeof(_Float16) * (long)B_ * D_ * CC_);
  float* retr = (float*)alloc(sizeof(float) * nBT_D);
  _Float16* yh = (_Float16*)alloc(sizeof(_Float16) * nBT_D);

  auto cast = [&](const float* in, _Float16* o, long n) {
    cast_f16<<<dim3((unsigned)((n + 255) / 256)), 256, 0, stream>>>(in, o, n);
  };
  cast(x, xh, nBT_D);
  cast(W_K, WKh, nDD);   cast(W_V, WVh, nDD);   cast(W_Q, WQh, nDD);
  cast(W_th, Wthh, nDD); cast(W_et, Weth, nDD); cast(W_al, Walh, nDD);
  cast(W_gate, Wgh, nDD); cast(W_proj, Wph, nDD);
  cast(W1, W1h, (long)DH_ * D_);

  zero_f32<<<dim3((unsigned)((3L * BT_ + 255) / 256)), 256, 0, stream>>>(accs, 3L * BT_);
  init_state<<<dim3((unsigned)((nState + 255) / 256)), 256, 0, stream>>>(W2, Sb, fWb, fW16b);

  // ---- projections: x @ W^T ----
  auto projF = [&](const _Float16* Wt, float* o) {
    GemmP p{}; p.A = xh; p.lda = D_; p.Bt = Wt; p.ldb = D_;
    p.M = BT_; p.N = D_; p.K = D_; p.out = o; p.ldo = D_;
    gemm_epi<0><<<dim3(D_ / 128, BT_ / 64, 1), 128, 0, stream>>>(p);
  };
  projF(WKh, kpre); projF(WVh, vpre); projF(WQh, qpre);

  auto projMean = [&](const _Float16* Wt, const float* bias, float* acc) {
    GemmP p{}; p.A = xh; p.lda = D_; p.Bt = Wt; p.ldb = D_;
    p.M = BT_; p.N = D_; p.K = D_;
    p.bias = bias; p.rowacc = acc; p.sRow = 0; p.invN = 1.0f / D_;
    gemm_epi<2><<<dim3(D_ / 128, BT_ / 64, 1), 128, 0, stream>>>(p);
  };
  projMean(Wthh, b_th, thetaA);
  projMean(Weth, b_et, etaA);
  projMean(Walh, b_al, alphaA);

  {
    GemmP p{}; p.A = xh; p.lda = D_; p.Bt = Wgh; p.ldb = D_;
    p.M = BT_; p.N = D_; p.K = D_; p.out = gateS; p.ldo = D_;
    gemm_epi<1><<<dim3(D_ / 128, BT_ / 64, 1), 128, 0, stream>>>(p);
  }

  // ---- depthwise conv (+ l2norm for k/q) ----
  conv_kernel<<<BT_, 256, 0, stream>>>(kpre, conv_k, nullptr, kh, 1);
  conv_kernel<<<BT_, 256, 0, stream>>>(qpre, conv_q, nullptr, qh, 1);
  conv_kernel<<<BT_, 256, 0, stream>>>(vpre, conv_v, vF, nullptr, 0);

  // ---- h_k = silu(k @ W1^T) (+ transposed copy), h_q = silu(q @ W1^T) ----
  {
    GemmP p{}; p.A = kh; p.lda = D_; p.Bt = W1h; p.ldb = D_;
    p.M = BT_; p.N = DH_; p.K = D_;
    p.out16 = hk; p.ldo16 = DH_; p.outT = hkT;
    gemm_epi<3><<<dim3(DH_ / 128, BT_ / 64, 1), 128, 0, stream>>>(p);
  }
  {
    GemmP p{}; p.A = qh; p.lda = D_; p.Bt = W1h; p.ldb = D_;
    p.M = BT_; p.N = DH_; p.K = D_;
    p.out16 = hq; p.ldo16 = DH_; p.outT = nullptr;
    gemm_epi<3><<<dim3(DH_ / 128, BT_ / 64, 1), 128, 0, stream>>>(p);
  }

  coef_kernel<<<B_ * NC_, CC_, 0, stream>>>(thetaA, etaA, alphaA, wSb, wFb, eCb, bCb, sSb);

  // ---- sequential chunk loop ----
  for (int c = 0; c <= NC_; ++c) {
    if (c > 0) {  // retrieval for chunk c-1 uses fW after its update (current state)
      const int cm = c - 1;
      GemmP p{};
      p.A = hq + (long)cm * CC_ * DH_; p.sA = (long)T_ * DH_; p.lda = DH_;
      p.Bt = fW16b; p.sB = (long)D_ * DH_; p.ldb = DH_;
      p.M = CC_; p.N = D_; p.K = DH_;
      p.out = retr + (long)cm * CC_ * D_; p.sO = (long)T_ * D_; p.ldo = D_;
      gemm_epi<0><<<dim3(D_ / 128, CC_ / 64, B_), 128, 0, stream>>>(p);
    }
    if (c < NC_) {
      {  // pred = hk_c @ fW^T  ->  scaled-residual f16 transposes
        GemmP p{};
        p.A = hk + (long)c * CC_ * DH_; p.sA = (long)T_ * DH_; p.lda = DH_;
        p.Bt = fW16b; p.sB = (long)D_ * DH_; p.ldb = DH_;
        p.M = CC_; p.N = D_; p.K = DH_;
        p.v = vF + (long)c * CC_ * D_; p.sV = (long)T_ * D_; p.ldv = D_;
        p.wS = wSb + (long)c * CC_; p.wF = wFb + (long)c * CC_; p.sW = (long)NC_ * CC_;
        p.r1T = r1Tb; p.r2T = r2Tb; p.sR = (long)D_ * CC_;
        gemm_epi<4><<<dim3(D_ / 128, CC_ / 64, B_), 128, 0, stream>>>(p);
      }
      state_update<<<dim3(DH_ / 64, D_ / 64, B_), 128, 0, stream>>>(
          r1Tb, r2Tb, (long)D_ * CC_,
          hkT, (long)DH_ * T_, c * CC_,
          Sb, fWb, fW16b,
          eCb + c, bCb + c, sSb + c, NC_);
    }
  }

  // ---- layernorm + gate, final projection ----
  ln_gate<<<BT_, 256, 0, stream>>>(retr, gateS, ln_g, ln_b, yh);
  {
    GemmP p{}; p.A = yh; p.lda = D_; p.Bt = Wph; p.ldb = D_;
    p.M = BT_; p.N = D_; p.K = D_;
    p.out = (float*)d_out; p.ldo = D_;
    gemm_epi<0><<<dim3(D_ / 128, BT_ / 64, 1), 128, 0, stream>>>(p);
  }
}
